// Model_23622320128521
// MI455X (gfx1250) — matrix operand order repaired and checked by reading, not compiled, not run on hardware
//
#include <hip/hip_runtime.h>

// ---------------------------------------------------------------------------
// Types / helpers
// ---------------------------------------------------------------------------
typedef __bf16 bf16;
typedef __attribute__((ext_vector_type(16))) __bf16 v16bf;
typedef __attribute__((ext_vector_type(8)))  float  v8f;

static __device__ __forceinline__ float b2f(bf16 x) {
    unsigned short u = __builtin_bit_cast(unsigned short, x);
    unsigned int w = ((unsigned int)u) << 16;
    return __builtin_bit_cast(float, w);
}
static __device__ __forceinline__ bf16 f2b(float f) {
    unsigned int u = __builtin_bit_cast(unsigned int, f);
    unsigned int r = u + 0x7FFFu + ((u >> 16) & 1u);   // round-to-nearest-even
    unsigned short s = (unsigned short)(r >> 16);
    return __builtin_bit_cast(bf16, s);
}
static __device__ __forceinline__ v8f v8zero() {
    v8f z;
#pragma unroll
    for (int i = 0; i < 8; ++i) z[i] = 0.f;
    return z;
}
union Frag { v16bf v; uint4 u[2]; };

static __device__ __forceinline__ v8f wmma_bf16(v16bf a, v16bf b, v8f c) {
    return __builtin_amdgcn_wmma_f32_16x16x32_bf16(false, a, false, b, (short)0, c, false, false);
}

// ---- CDNA5 async global->LDS copy (ASYNCcnt-tracked), with sync fallback ----
#if defined(__gfx1250__) && __has_builtin(__builtin_amdgcn_global_load_async_to_lds_b128)
#define HAS_ASYNC_LDS 1
#else
#define HAS_ASYNC_LDS 0
#endif

#if HAS_ASYNC_LDS
typedef int i4v __attribute__((vector_size(4 * sizeof(int))));
typedef __attribute__((address_space(1))) i4v* gl_i4p;
typedef __attribute__((address_space(3))) i4v* lds_i4p;
typedef __attribute__((address_space(1))) void* gl_vp;
typedef __attribute__((address_space(3))) void* lds_vp;
#endif

static __device__ __forceinline__ void async_cp16(const bf16* g, bf16* l) {
#if HAS_ASYNC_LDS
    __builtin_amdgcn_global_load_async_to_lds_b128(
        (gl_i4p)(gl_vp)(void*)const_cast<bf16*>(g),
        (lds_i4p)(lds_vp)(void*)l,
        0, 0);
#else
    *(uint4*)l = *(const uint4*)g;
#endif
}

#if HAS_ASYNC_LDS
#if __has_builtin(__builtin_amdgcn_s_wait_asynccnt)
#define WAIT_ASYNC(N) __builtin_amdgcn_s_wait_asynccnt(N)
#else
#define WAIT_ASYNC(N) asm volatile("s_wait_asynccnt %0" ::"i"(N) : "memory")
#endif
#else
#define WAIT_ASYNC(N) ((void)0)
#endif

// Problem constants
#define PS   6
#define BB   16
#define CC   8
#define LL   64
#define DD   512
#define DFF  2048
#define PRED 96
#define MM   (BB*CC*LL)   // 8192 tokens per expert
#define HH   8
#define DH   64

// ---------------------------------------------------------------------------
// Weight convert+transpose: W (K x N) f32  ->  Wt (N x K) bf16
// ---------------------------------------------------------------------------
__global__ __launch_bounds__(256) void cvt_transpose_kernel(const float* __restrict__ W,
                                                            bf16* __restrict__ Wt,
                                                            int K, int N) {
    int idx = blockIdx.x * 256 + threadIdx.x;
    if (idx >= K * N) return;
    int k = idx / N, n = idx % N;
    Wt[(size_t)n * K + k] = f2b(W[idx]);
}

// Copy one expert slice to f32 residual + bf16 activation
__global__ __launch_bounds__(256) void cvt_expert_kernel(const float* __restrict__ src,
                                                         float* __restrict__ X,
                                                         bf16* __restrict__ Xb) {
    int idx = blockIdx.x * 256 + threadIdx.x;
    if (idx >= MM * DD) return;
    float v = src[idx];
    X[idx]  = v;
    Xb[idx] = f2b(v);
}

// ---------------------------------------------------------------------------
// GEMM: out = act(A(MxK,bf16) @ B(KxN) + bias), B given transposed Bt(N x K)
// Tile 128x128x32; 256 threads = 8 waves; each wave 64x32 (4x2 wmma frags).
// Double-buffered LDS staging via async global->LDS copies.
// ---------------------------------------------------------------------------
__global__ __launch_bounds__(256) void gemm_bf16_kernel(const bf16* __restrict__ A,
                                                        const bf16* __restrict__ Bt,
                                                        const float* __restrict__ bias,
                                                        float* __restrict__ outf,
                                                        bf16* __restrict__ outb,
                                                        int M, int N, int K, int relu) {
    __shared__ __align__(16) bf16 As[2][128][40];   // 80B pitch -> conflict-free frags
    __shared__ __align__(16) bf16 Bs[2][128][40];

    const int tid  = threadIdx.x;
    const int lane = tid & 31;
    const int wid  = tid >> 5;
    const int wm = (wid >> 2) * 64;     // wave M offset in tile
    const int wn = (wid & 3)  * 32;     // wave N offset in tile
    const int nl = lane & 15;
    const int kg = lane >> 4;

    const int tile_m = blockIdx.y * 128;
    const int tile_n = blockIdx.x * 128;

    // per-thread staging chunk coordinates (2 chunks per operand per thread)
    const int r0 = tid >> 2,           o0 = (tid & 3) * 8;            // chunk tid
    const int r1 = (tid + 256) >> 2,   o1 = ((tid + 256) & 3) * 8;    // chunk tid+256

    const bf16* Arow0 = A  + (size_t)(tile_m + r0) * K + o0;
    const bf16* Arow1 = A  + (size_t)(tile_m + r1) * K + o1;
    const bf16* Brow0 = Bt + (size_t)(tile_n + r0) * K + o0;
    const bf16* Brow1 = Bt + (size_t)(tile_n + r1) * K + o1;

    v8f acc[4][2];
#pragma unroll
    for (int i = 0; i < 4; ++i)
#pragma unroll
        for (int j = 0; j < 2; ++j) acc[i][j] = v8zero();

    const int nsteps = K >> 5;   // K / 32

    // prologue: stage tile 0 into buffer 0
    async_cp16(Arow0, &As[0][r0][o0]);
    async_cp16(Arow1, &As[0][r1][o1]);
    async_cp16(Brow0, &Bs[0][r0][o0]);
    async_cp16(Brow1, &Bs[0][r1][o1]);

    for (int s = 0; s < nsteps; ++s) {
        const int buf = s & 1;
        const int k1  = (s + 1) << 5;
        if (s + 1 < nsteps) {
            // stage next tile into the other buffer (async, overlaps compute)
            async_cp16(Arow0 + k1, &As[buf ^ 1][r0][o0]);
            async_cp16(Arow1 + k1, &As[buf ^ 1][r1][o1]);
            async_cp16(Brow0 + k1, &Bs[buf ^ 1][r0][o0]);
            async_cp16(Brow1 + k1, &Bs[buf ^ 1][r1][o1]);
            WAIT_ASYNC(4);       // in-order: current tile's 4 copies are done
        } else {
            WAIT_ASYNC(0);
        }
        __syncthreads();         // all waves' copies for `buf` visible

        Frag a[4], b[2];
#pragma unroll
        for (int i = 0; i < 4; ++i) {
            int row = wm + 16 * i + nl;
            a[i].u[0] = *(const uint4*)&As[buf][row][kg * 8];
            a[i].u[1] = *(const uint4*)&As[buf][row][16 + kg * 8];
        }
#pragma unroll
        for (int j = 0; j < 2; ++j) {
            int row = wn + 16 * j + nl;
            b[j].u[0] = *(const uint4*)&Bs[buf][row][kg * 8];
            b[j].u[1] = *(const uint4*)&Bs[buf][row][16 + kg * 8];
        }
#pragma unroll
        for (int i = 0; i < 4; ++i)
#pragma unroll
            for (int j = 0; j < 2; ++j)
                acc[i][j] = wmma_bf16(a[i].v, b[j].v, acc[i][j]);

        __syncthreads();         // protect `buf` from next iteration's staging
    }

    // ---- epilogue ----
#pragma unroll
    for (int i = 0; i < 4; ++i)
#pragma unroll
        for (int j = 0; j < 2; ++j)
#pragma unroll
            for (int v = 0; v < 8; ++v) {
                int m = tile_m + wm + 16 * i + 8 * kg + v;
                int n = tile_n + wn + 16 * j + nl;
                float val = acc[i][j][v] + bias[n];
                if (relu) val = fmaxf(val, 0.f);
                size_t o = (size_t)m * N + n;
                if (outf) outf[o] = val;
                if (outb) outb[o] = f2b(val);
            }
}

// ---------------------------------------------------------------------------
// Attention #1: per (b,c,h), S=64, dh=64, fully WMMA. One wave per head.
// ---------------------------------------------------------------------------
__global__ __launch_bounds__(32) void attn1_kernel(const bf16* __restrict__ Qm,
                                                   const bf16* __restrict__ Km,
                                                   const bf16* __restrict__ Vm,
                                                   bf16* __restrict__ Om) {
    __shared__ __align__(16) bf16 P[64][72];    // softmaxed probs (bf16)
    __shared__ __align__(16) bf16 Vt[64][72];   // V transposed: Vt[d][s]

    const int g = blockIdx.x;
    const int h = g & 7;
    const int c = (g >> 3) & 7;
    const int b = g >> 6;
    const size_t tok0 = (size_t)(b * CC + c) * LL;

    const int lane = threadIdx.x;
    const int nl = lane & 15;
    const int kg = lane >> 4;

    // ---- scores = Q @ K^T ----
    v8f s[4][4];
#pragma unroll
    for (int i = 0; i < 4; ++i)
#pragma unroll
        for (int j = 0; j < 4; ++j) s[i][j] = v8zero();

#pragma unroll
    for (int kt = 0; kt < 2; ++kt) {
        Frag aq[4], bk[4];
#pragma unroll
        for (int i = 0; i < 4; ++i) {
            const bf16* p = Qm + (tok0 + 16 * i + nl) * DD + h * DH + kt * 32;
            aq[i].u[0] = *(const uint4*)(p + kg * 8);
            aq[i].u[1] = *(const uint4*)(p + 16 + kg * 8);
        }
#pragma unroll
        for (int j = 0; j < 4; ++j) {
            const bf16* p = Km + (tok0 + 16 * j + nl) * DD + h * DH + kt * 32 + kg * 8;
            bk[j].u[0] = *(const uint4*)p;
            bk[j].u[1] = *(const uint4*)(p + 16);
        }
#pragma unroll
        for (int i = 0; i < 4; ++i)
#pragma unroll
            for (int j = 0; j < 4; ++j)
                s[i][j] = wmma_bf16(aq[i].v, bk[j].v, s[i][j]);
    }

    // ---- softmax over 64 cols (4 j-tiles x 16 lanes in row-group) ----
    const float scale = 0.125f;   // 1/sqrt(64)
#pragma unroll
    for (int i = 0; i < 4; ++i) {
#pragma unroll
        for (int v = 0; v < 8; ++v) {
            float mx = -3.0e38f;
#pragma unroll
            for (int j = 0; j < 4; ++j) mx = fmaxf(mx, s[i][j][v]);
            for (int o = 1; o < 16; o <<= 1) mx = fmaxf(mx, __shfl_xor(mx, o));
            float sum = 0.f;
#pragma unroll
            for (int j = 0; j < 4; ++j) {
                float e = __expf((s[i][j][v] - mx) * scale);
                s[i][j][v] = e;
                sum += e;
            }
            for (int o = 1; o < 16; o <<= 1) sum += __shfl_xor(sum, o);
            float rinv = 1.f / sum;
            int m = 16 * i + 8 * kg + v;
#pragma unroll
            for (int j = 0; j < 4; ++j)
                P[m][16 * j + nl] = f2b(s[i][j][v] * rinv);
        }
    }

    // ---- stage V transposed ----
#pragma unroll
    for (int r = 0; r < 2; ++r) {
        int srow = lane + 32 * r;
        const bf16* vp = Vm + (tok0 + srow) * DD + h * DH;
#pragma unroll
        for (int d = 0; d < 64; ++d) Vt[d][srow] = vp[d];
    }
    __syncthreads();

    // ---- O = P @ V ----
    v8f o[4][4];
#pragma unroll
    for (int i = 0; i < 4; ++i)
#pragma unroll
        for (int j = 0; j < 4; ++j) o[i][j] = v8zero();

#pragma unroll
    for (int kt = 0; kt < 2; ++kt) {
        Frag ap[4], bv[4];
#pragma unroll
        for (int i = 0; i < 4; ++i) {
            const bf16* p = &P[16 * i + nl][kt * 32];
            ap[i].u[0] = *(const uint4*)(p + kg * 8);
            ap[i].u[1] = *(const uint4*)(p + 16 + kg * 8);
        }
#pragma unroll
        for (int j = 0; j < 4; ++j) {
            const bf16* p = &Vt[16 * j + nl][kt * 32 + kg * 8];
            bv[j].u[0] = *(const uint4*)p;
            bv[j].u[1] = *(const uint4*)(p + 16);
        }
#pragma unroll
        for (int i = 0; i < 4; ++i)
#pragma unroll
            for (int j = 0; j < 4; ++j)
                o[i][j] = wmma_bf16(ap[i].v, bv[j].v, o[i][j]);
    }

#pragma unroll
    for (int i = 0; i < 4; ++i)
#pragma unroll
        for (int j = 0; j < 4; ++j)
#pragma unroll
            for (int v = 0; v < 8; ++v) {
                int m = 16 * i + 8 * kg + v;
                int n = 16 * j + nl;
                Om[(tok0 + m) * DD + h * DH + n] = f2b(o[i][j][v]);
            }
}

// ---------------------------------------------------------------------------
// Attention #2: per (b,l,h,sq), S=8 over chunk axis — VALU (0.03% of FLOPs)
// ---------------------------------------------------------------------------
__global__ __launch_bounds__(256) void attn2_kernel(const bf16* __restrict__ Qm,
                                                    const bf16* __restrict__ Km,
                                                    const bf16* __restrict__ Vm,
                                                    bf16* __restrict__ Om) {
    int t = blockIdx.x * 256 + threadIdx.x;   // 65536 threads
    int sq = t & 7;
    int h  = (t >> 3) & 7;
    int l  = (t >> 6) & 63;
    int b  = t >> 12;

    float qv[64];
    {
        const bf16* qp = Qm + ((size_t)(b * CC + sq) * LL + l) * DD + h * DH;
#pragma unroll
        for (int d = 0; d < 64; ++d) qv[d] = b2f(qp[d]);
    }
    float sc[8];
#pragma unroll
    for (int c = 0; c < 8; ++c) {
        const bf16* kp = Km + ((size_t)(b * CC + c) * LL + l) * DD + h * DH;
        float a = 0.f;
#pragma unroll
        for (int d = 0; d < 64; ++d) a += qv[d] * b2f(kp[d]);
        sc[c] = a * 0.125f;
    }
    float mx = sc[0];
#pragma unroll
    for (int c = 1; c < 8; ++c) mx = fmaxf(mx, sc[c]);
    float sum = 0.f;
#pragma unroll
    for (int c = 0; c < 8; ++c) { sc[c] = __expf(sc[c] - mx); sum += sc[c]; }
    float rinv = 1.f / sum;

    bf16* op = Om + ((size_t)(b * CC + sq) * LL + l) * DD + h * DH;
#pragma unroll
    for (int d = 0; d < 64; ++d) {
        float a = 0.f;
#pragma unroll
        for (int c = 0; c < 8; ++c)
            a += sc[c] * b2f(Vm[((size_t)(b * CC + c) * LL + l) * DD + h * DH + d]);
        op[d] = f2b(a * rinv);
    }
}

// ---------------------------------------------------------------------------
// LayerNorm: X = LN(X + delta)*g + b  (in place), and Xb = bf16(X)
// ---------------------------------------------------------------------------
__global__ __launch_bounds__(256) void ln_kernel(float* __restrict__ X,
                                                 const float* __restrict__ delta,
                                                 const float* __restrict__ g,
                                                 const float* __restrict__ bta,
                                                 bf16* __restrict__ Xb) {
    __shared__ float rs[256], rq[256];
    const int row = blockIdx.x;
    const int tid = threadIdx.x;
    const size_t base = (size_t)row * DD;

    float x0 = X[base + tid]       + delta[base + tid];
    float x1 = X[base + tid + 256] + delta[base + tid + 256];
    rs[tid] = x0 + x1;
    rq[tid] = x0 * x0 + x1 * x1;
    __syncthreads();
    for (int o = 128; o > 0; o >>= 1) {
        if (tid < o) { rs[tid] += rs[tid + o]; rq[tid] += rq[tid + o]; }
        __syncthreads();
    }
    float mean = rs[0] * (1.f / DD);
    float var  = rq[0] * (1.f / DD) - mean * mean;
    float rstd = rsqrtf(var + 1e-5f);

    float y0 = (x0 - mean) * rstd * g[tid]       + bta[tid];
    float y1 = (x1 - mean) * rstd * g[tid + 256] + bta[tid + 256];
    X[base + tid]        = y0;
    X[base + tid + 256]  = y1;
    Xb[base + tid]       = f2b(y0);
    Xb[base + tid + 256] = f2b(y1);
}

// Gather rows l==63 into lastbuf[p][b*8+c][512]
__global__ __launch_bounds__(256) void gather_last_kernel(const float* __restrict__ X,
                                                          float* __restrict__ last,
                                                          int p) {
    int idx = blockIdx.x * 256 + threadIdx.x;   // 128*512
    if (idx >= 128 * DD) return;
    int r = idx >> 9, d = idx & 511;
    last[((size_t)p * 128 + r) * DD + d] = X[((size_t)r * LL + 63) * DD + d];
}

// Gated combine + head projection + transpose: out[b][j][c]
__global__ __launch_bounds__(128) void final_head_kernel(const float* __restrict__ last,
                                                         const float* __restrict__ gates,
                                                         const float* __restrict__ hW,
                                                         const float* __restrict__ hb,
                                                         float* __restrict__ out) {
    __shared__ float comb[DD];
    const int bc = blockIdx.x;       // 0..127
    const int b = bc >> 3, c = bc & 7;
    const int tid = threadIdx.x;

    for (int d = tid; d < DD; d += 128) {
        float s = 0.f;
#pragma unroll
        for (int p = 0; p < PS; ++p)
            s += gates[b * PS + p] * last[((size_t)p * 128 + bc) * DD + d];
        comb[d] = s;
    }
    __syncthreads();
    if (tid < PRED) {
        float acc = hb[tid];
        for (int d = 0; d < DD; ++d) acc += comb[d] * hW[(size_t)d * PRED + tid];
        out[((size_t)b * PRED + tid) * CC + c] = acc;
    }
}

// ---------------------------------------------------------------------------
// Host launcher
// ---------------------------------------------------------------------------
extern "C" void kernel_launch(void* const* d_in, const int* in_sizes, int n_in,
                              void* d_out, int out_size, void* d_ws, size_t ws_size,
                              hipStream_t stream) {
    const float* expert_x = (const float*)d_in[0];
    const float* gates    = (const float*)d_in[1];
    const float* cWq = (const float*)d_in[2];  const float* cbq = (const float*)d_in[3];
    const float* cWk = (const float*)d_in[4];  const float* cbk = (const float*)d_in[5];
    const float* cWv = (const float*)d_in[6];  const float* cbv = (const float*)d_in[7];
    const float* cWo = (const float*)d_in[8];  const float* cbo = (const float*)d_in[9];
    const float* iWq = (const float*)d_in[10]; const float* ibq = (const float*)d_in[11];
    const float* iWk = (const float*)d_in[12]; const float* ibk = (const float*)d_in[13];
    const float* iWv = (const float*)d_in[14]; const float* ibv = (const float*)d_in[15];
    const float* iWo = (const float*)d_in[16]; const float* ibo = (const float*)d_in[17];
    const float* mW1 = (const float*)d_in[18]; const float* mb1 = (const float*)d_in[19];
    const float* mW2 = (const float*)d_in[20]; const float* mb2 = (const float*)d_in[21];
    const float* g1 = (const float*)d_in[22];  const float* b1 = (const float*)d_in[23];
    const float* g3 = (const float*)d_in[24];  const float* b3 = (const float*)d_in[25];
    const float* g4 = (const float*)d_in[26];  const float* b4 = (const float*)d_in[27];
    const float* hW = (const float*)d_in[28];  const float* hb = (const float*)d_in[29];

    char* ws = (char*)d_ws;
    // workspace layout (bytes)
    bf16* wt_cq = (bf16*)(ws + 0);
    bf16* wt_ck = (bf16*)(ws + 524288);
    bf16* wt_cv = (bf16*)(ws + 1048576);
    bf16* wt_co = (bf16*)(ws + 1572864);
    bf16* wt_iq = (bf16*)(ws + 2097152);
    bf16* wt_ik = (bf16*)(ws + 2621440);
    bf16* wt_iv = (bf16*)(ws + 3145728);
    bf16* wt_io = (bf16*)(ws + 3670016);
    bf16* wt_m1 = (bf16*)(ws + 4194304);   // 2048 x 512
    bf16* wt_m2 = (bf16*)(ws + 6291456);   // 512 x 2048
    float* X    = (float*)(ws + 8388608);  // 8192 x 512 f32
    bf16*  Xb   = (bf16*)(ws + 25165824);
    bf16*  Qb   = (bf16*)(ws + 33554432);
    bf16*  Kb   = (bf16*)(ws + 41943040);
    bf16*  Vb   = (bf16*)(ws + 50331648);
    bf16*  Ob   = (bf16*)(ws + 58720256);
    float* Pf   = (float*)(ws + 67108864); // 8192 x 512 f32 (gemm f32 out)
    bf16*  H1   = (bf16*)(ws + 83886080);  // 8192 x 2048
    float* last = (float*)(ws + 117440512);

    // ---- weights: convert + transpose to bf16 ----
    {
        int n512 = (512 * 512 + 255) / 256;
        cvt_transpose_kernel<<<n512, 256, 0, stream>>>(cWq, wt_cq, 512, 512);
        cvt_transpose_kernel<<<n512, 256, 0, stream>>>(cWk, wt_ck, 512, 512);
        cvt_transpose_kernel<<<n512, 256, 0, stream>>>(cWv, wt_cv, 512, 512);
        cvt_transpose_kernel<<<n512, 256, 0, stream>>>(cWo, wt_co, 512, 512);
        cvt_transpose_kernel<<<n512, 256, 0, stream>>>(iWq, wt_iq, 512, 512);
        cvt_transpose_kernel<<<n512, 256, 0, stream>>>(iWk, wt_ik, 512, 512);
        cvt_transpose_kernel<<<n512, 256, 0, stream>>>(iWv, wt_iv, 512, 512);
        cvt_transpose_kernel<<<n512, 256, 0, stream>>>(iWo, wt_io, 512, 512);
        int nmlp = (512 * 2048 + 255) / 256;
        cvt_transpose_kernel<<<nmlp, 256, 0, stream>>>(mW1, wt_m1, 512, 2048);
        cvt_transpose_kernel<<<nmlp, 256, 0, stream>>>(mW2, wt_m2, 2048, 512);
    }

    const dim3 gemm_blk(256);
    const dim3 g512(512 / 128, MM / 128);    // N=512
    const dim3 g2048(2048 / 128, MM / 128);  // N=2048

    for (int p = 0; p < PS; ++p) {
        const float* xp = expert_x + (size_t)p * MM * DD;
        cvt_expert_kernel<<<(MM * DD + 255) / 256, 256, 0, stream>>>(xp, X, Xb);

        // chunk-attention QKV
        gemm_bf16_kernel<<<g512, gemm_blk, 0, stream>>>(Xb, wt_cq, cbq, nullptr, Qb, MM, 512, 512, 0);
        gemm_bf16_kernel<<<g512, gemm_blk, 0, stream>>>(Xb, wt_ck, cbk, nullptr, Kb, MM, 512, 512, 0);
        gemm_bf16_kernel<<<g512, gemm_blk, 0, stream>>>(Xb, wt_cv, cbv, nullptr, Vb, MM, 512, 512, 0);
        attn1_kernel<<<BB * CC * HH, 32, 0, stream>>>(Qb, Kb, Vb, Ob);
        gemm_bf16_kernel<<<g512, gemm_blk, 0, stream>>>(Ob, wt_co, cbo, Pf, nullptr, MM, 512, 512, 0);
        ln_kernel<<<MM, 256, 0, stream>>>(X, Pf, g1, b1, Xb);

        // inter-chunk attention QKV
        gemm_bf16_kernel<<<g512, gemm_blk, 0, stream>>>(Xb, wt_iq, ibq, nullptr, Qb, MM, 512, 512, 0);
        gemm_bf16_kernel<<<g512, gemm_blk, 0, stream>>>(Xb, wt_ik, ibk, nullptr, Kb, MM, 512, 512, 0);
        gemm_bf16_kernel<<<g512, gemm_blk, 0, stream>>>(Xb, wt_iv, ibv, nullptr, Vb, MM, 512, 512, 0);
        attn2_kernel<<<(BB * LL * HH * CC) / 256, 256, 0, stream>>>(Qb, Kb, Vb, Ob);
        gemm_bf16_kernel<<<g512, gemm_blk, 0, stream>>>(Ob, wt_io, ibo, Pf, nullptr, MM, 512, 512, 0);
        ln_kernel<<<MM, 256, 0, stream>>>(X, Pf, g3, b3, Xb);

        // MLP
        gemm_bf16_kernel<<<g2048, gemm_blk, 0, stream>>>(Xb, wt_m1, mb1, nullptr, H1, MM, 2048, 512, 1);
        gemm_bf16_kernel<<<g512, gemm_blk, 0, stream>>>(H1, wt_m2, mb2, Pf, nullptr, MM, 512, 2048, 0);
        ln_kernel<<<MM, 256, 0, stream>>>(X, Pf, g4, b4, Xb);

        gather_last_kernel<<<(128 * DD + 255) / 256, 256, 0, stream>>>(X, last, p);
    }

    final_head_kernel<<<128, 128, 0, stream>>>(last, gates, hW, hb, (float*)d_out);
}
